// SelfAttention_1417339207749
// MI455X (gfx1250) — matrix operand order repaired and checked by reading, not compiled, not run on hardware
//
#include <hip/hip_runtime.h>
#include <hip/hip_bf16.h>

typedef _Float16 half8  __attribute__((ext_vector_type(8)));
typedef _Float16 half16 __attribute__((ext_vector_type(16)));
typedef float    float8 __attribute__((ext_vector_type(8)));

#define BB 4
#define TT 2048
#define CC 768
#define HH 64

static __device__ __forceinline__ half16 combine16(half8 lo, half8 hi) {
  return __builtin_shufflevector(lo, hi, 0,1,2,3,4,5,6,7,8,9,10,11,12,13,14,15);
}

// A fragment (16-bit, 16xK, row-major source). Lane l: row M=l%16;
// halves 0..7 = K[k0+8g .. +7], halves 8..15 = K[k0+16+8g .. +7], g=l/16.
static __device__ __forceinline__ half16 load_a_frag(const _Float16* base, int ld,
                                                     int lane, int k0) {
  const int row = lane & 15, grp = lane >> 4;
  const _Float16* p = base + (size_t)row * ld + k0 + grp * 8;
  half8 lo = *(const half8*)(p);
  half8 hi = *(const half8*)(p + 16);
  return combine16(lo, hi);
}

// B fragment (16-bit, Kx16 logical, stored N-major [N][K] i.e. B^T rows).
// Lane l: column N=l%16; halves 0..7 = K[k0+8g .. +7], halves 8..15 = K[k0+16+8g .. +7].
static __device__ __forceinline__ half16 load_b_frag(const _Float16* baseNT, int ld,
                                                     int lane, int k0) {
  const int n = lane & 15, grp = lane >> 4;
  const _Float16* p = baseNT + (size_t)n * ld + k0 + grp * 8;
  half8 lo = *(const half8*)(p);
  half8 hi = *(const half8*)(p + 16);
  return combine16(lo, hi);
}

// ---------------------------------------------------------------------------
// Kernel 1: fp32 -> f16 conversion. x stays row-major [B,T,C]; weights are
// transposed to [3][H][C] (N-major B-fragment source).
// ---------------------------------------------------------------------------
__global__ __launch_bounds__(256) void convert_kernel(
    const float* __restrict__ x, const float* __restrict__ Wq,
    const float* __restrict__ Wk, const float* __restrict__ Wv,
    _Float16* __restrict__ xh, _Float16* __restrict__ WT) {
  const long nX = (long)BB * TT * CC;
  const long nW = 3L * HH * CC;
  const long total = nX + nW;
  for (long i = (long)blockIdx.x * blockDim.x + threadIdx.x; i < total;
       i += (long)gridDim.x * blockDim.x) {
    if (i < nX) {
      xh[i] = (_Float16)x[i];
    } else {
      long j = i - nX;                 // target index in WT[3][H][C]
      int proj = (int)(j / (HH * CC));
      int r    = (int)(j % (HH * CC));
      int h = r / CC, c = r % CC;
      const float* W = (proj == 0) ? Wq : (proj == 1) ? Wk : Wv;
      WT[j] = (_Float16)W[(size_t)c * HH + h];   // W is [C,H] -> WT is [H,C]
    }
  }
}

// ---------------------------------------------------------------------------
// Kernel 2: q/k/v projections. One wave = one 16(time) x 16(head-dim) tile;
// K=768 in 24 v_wmma_f32_16x16x32_f16 steps. q,k -> f16 [B,T,H]; v -> f16
// transposed [B,H,T]. Tile indices pinned to SGPRs via readfirstlane.
// ---------------------------------------------------------------------------
__global__ __launch_bounds__(256) void proj_kernel(
    const _Float16* __restrict__ xh, const _Float16* __restrict__ WT,
    _Float16* __restrict__ qh, _Float16* __restrict__ kh,
    _Float16* __restrict__ vT) {
  const int lane = threadIdx.x & 31;
  const int tile = __builtin_amdgcn_readfirstlane(blockIdx.x * 8 + (threadIdx.x >> 5));
  const int b    = tile / (12 * (TT / 16));
  int rem        = tile % (12 * (TT / 16));
  const int t16  = rem / 12;
  rem            = rem % 12;
  const int proj = rem >> 2;                    // 0=q 1=k 2=v (SGPR-uniform)
  const int nt   = rem & 3;                     // 16-wide slice of H=64

  const _Float16* Xbase = xh + (size_t)b * TT * CC + (size_t)(t16 * 16) * CC;
  const _Float16* Wbase = WT + (size_t)proj * HH * CC + (size_t)(nt * 16) * CC;

  float8 acc = {};
#pragma unroll 4
  for (int kk = 0; kk < CC / 32; ++kk) {
    const int k0 = kk * 32;
    half16 a = load_a_frag(Xbase, CC, lane, k0);
    half16 w = load_b_frag(Wbase, CC, lane, k0);
    acc = __builtin_amdgcn_wmma_f32_16x16x32_f16(false, a, false, w,
                                                 (short)0, acc, false, false);
  }

  const int grp = lane >> 4, col = lane & 15;
  const int h = nt * 16 + col;
  if (proj == 2) {                              // scalar branch (proj in SGPR)
    _Float16* dst = vT + (size_t)b * HH * TT + (size_t)h * TT;
#pragma unroll
    for (int r = 0; r < 8; ++r) dst[t16 * 16 + r + 8 * grp] = (_Float16)acc[r];
  } else {
    _Float16* dst = (proj == 0 ? qh : kh) + (size_t)b * TT * HH + h;
#pragma unroll
    for (int r = 0; r < 8; ++r)
      dst[(size_t)(t16 * 16 + r + 8 * grp) * HH] = (_Float16)acc[r];
  }
}

// ---------------------------------------------------------------------------
// Kernel 3: causal flash attention. One wave = one 16-query tile; 64-key
// blocks. 18 WMMAs per block: 8 for S, 8 for P*V, 2 for l = P*1 (row-sum on
// the tensor core -> no shuffle reduction for l).
// ---------------------------------------------------------------------------
__global__ __launch_bounds__(128) void attn_kernel(
    const _Float16* __restrict__ qh, const _Float16* __restrict__ kh,
    const _Float16* __restrict__ vT, float* __restrict__ out) {
  __shared__ alignas(16) _Float16 Plds[4][16 * 64];   // per-wave P tile (16x64)

  const int lane = threadIdx.x & 31;
  const int warp = threadIdx.x >> 5;
  const int grp = lane >> 4, col = lane & 15;
  const int qt = __builtin_amdgcn_readfirstlane(blockIdx.x * 4 + warp);
  const int b  = qt / (TT / 16);
  const int qi = qt % (TT / 16);
  const int qrow0 = qi * 16;

  const _Float16* Q = qh + (size_t)b * TT * HH;
  const _Float16* K = kh + (size_t)b * TT * HH;
  const _Float16* V = vT + (size_t)b * HH * TT;
  const float scale = 0.03608439182435161f;     // 768^-0.5

  // Q fragments reused for every key block (K-dim = H = 64 -> 2 steps of 32).
  half16 Aq0 = load_a_frag(Q + (size_t)qrow0 * HH, HH, lane, 0);
  half16 Aq1 = load_a_frag(Q + (size_t)qrow0 * HH, HH, lane, 32);

  half16 Bones;                                  // all-ones B: l = P @ 1
#pragma unroll
  for (int i = 0; i < 16; ++i) Bones[i] = (_Float16)1.0f;

  float m[8], alpha[8], mnew[8];
  float8 acc[4], accl;
#pragma unroll
  for (int e = 0; e < 8; ++e) m[e] = -1e30f;
#pragma unroll
  for (int n = 0; n < 4; ++n) acc[n] = (float8){};
  accl = (float8){};

  const int jbmax = (qrow0 + 15) >> 6;          // inclusive, 64-key blocks
  for (int jb = 0; jb <= jbmax; ++jb) {
    const int key0 = jb * 64;

    // ---- S = Q K^T, 16x64 slab: 4 n-tiles x 2 K-steps ----
    float8 S[4];
#pragma unroll
    for (int n = 0; n < 4; ++n) {
      const _Float16* kp = K + (size_t)(key0 + 16 * n) * HH;
      half16 bK0 = load_b_frag(kp, HH, lane, 0);
      half16 bK1 = load_b_frag(kp, HH, lane, 32);
      float8 s = (float8){};
      s = __builtin_amdgcn_wmma_f32_16x16x32_f16(false, Aq0, false, bK0, (short)0, s, false, false);
      s = __builtin_amdgcn_wmma_f32_16x16x32_f16(false, Aq1, false, bK1, (short)0, s, false, false);
      S[n] = s;
    }

    // ---- scale + causal mask (wave-uniform predicate, per-lane selects) ----
    const bool partial = (key0 + 63) > qrow0;
#pragma unroll
    for (int e = 0; e < 8; ++e) {
      const int qrow = qrow0 + e + 8 * grp;
#pragma unroll
      for (int n = 0; n < 4; ++n) {
        float s = S[n][e] * scale;
        if (partial && (key0 + 16 * n + col > qrow)) s = -1e30f;
        S[n][e] = s;
      }
    }

    // ---- online softmax: row max (16-lane xor reduce), exp, stage P in LDS
#pragma unroll
    for (int e = 0; e < 8; ++e) {
      float mm = fmaxf(fmaxf(S[0][e], S[1][e]), fmaxf(S[2][e], S[3][e]));
      mm = fmaxf(mm, __shfl_xor(mm, 1, 32));
      mm = fmaxf(mm, __shfl_xor(mm, 2, 32));
      mm = fmaxf(mm, __shfl_xor(mm, 4, 32));
      mm = fmaxf(mm, __shfl_xor(mm, 8, 32));
      mnew[e] = fmaxf(m[e], mm);
      const int M = e + 8 * grp;
#pragma unroll
      for (int n = 0; n < 4; ++n) {
        float p = __expf(S[n][e] - mnew[e]);
        Plds[warp][M * 64 + 16 * n + col] = (_Float16)p;
      }
    }
#pragma unroll
    for (int e = 0; e < 8; ++e) {
      alpha[e] = __expf(m[e] - mnew[e]);
      m[e] = mnew[e];
    }
#pragma unroll
    for (int n = 0; n < 4; ++n)
#pragma unroll
      for (int e = 0; e < 8; ++e) acc[n][e] *= alpha[e];
#pragma unroll
    for (int e = 0; e < 8; ++e) accl[e] *= alpha[e];

    // intra-wave cross-lane handoff through LDS
    asm volatile("s_wait_dscnt 0x0" ::: "memory");

    // ---- P re-loaded as two A fragments (16x64, ld=64) ----
    const _Float16* pbase = &Plds[warp][0];
    half16 Ap0 = load_a_frag(pbase, 64, lane, 0);
    half16 Ap1 = load_a_frag(pbase, 64, lane, 32);

    // l += P @ 1  (row-sum on the tensor core)
    accl = __builtin_amdgcn_wmma_f32_16x16x32_f16(false, Ap0, false, Bones, (short)0, accl, false, false);
    accl = __builtin_amdgcn_wmma_f32_16x16x32_f16(false, Ap1, false, Bones, (short)0, accl, false, false);

    // O += P V : V^T rows are B fragments, K = 64 keys in 2 steps
#pragma unroll
    for (int n = 0; n < 4; ++n) {
      const _Float16* vp = V + (size_t)(n * 16) * TT;
      half16 Bv0 = load_b_frag(vp, TT, lane, key0);
      half16 Bv1 = load_b_frag(vp, TT, lane, key0 + 32);
      acc[n] = __builtin_amdgcn_wmma_f32_16x16x32_f16(false, Ap0, false, Bv0, (short)0, acc[n], false, false);
      acc[n] = __builtin_amdgcn_wmma_f32_16x16x32_f16(false, Ap1, false, Bv1, (short)0, acc[n], false, false);
    }
    // order this iteration's LDS reads before next iteration's stores
    asm volatile("" ::: "memory");
  }

  // ---- epilogue: out = acc / l (each lane already holds its row's l) ----
#pragma unroll
  for (int n = 0; n < 4; ++n) {
    const int h = n * 16 + col;
#pragma unroll
    for (int e = 0; e < 8; ++e) {
      const int t = qrow0 + e + 8 * grp;
      out[(size_t)b * TT * HH + (size_t)t * HH + h] = acc[n][e] / accl[e];
    }
  }
}

// ---------------------------------------------------------------------------
extern "C" void kernel_launch(void* const* d_in, const int* in_sizes, int n_in,
                              void* d_out, int out_size, void* d_ws, size_t ws_size,
                              hipStream_t stream) {
  const float* x  = (const float*)d_in[0];
  const float* Wq = (const float*)d_in[1];
  const float* Wk = (const float*)d_in[2];
  const float* Wv = (const float*)d_in[3];
  float* out = (float*)d_out;

  char* ws = (char*)d_ws;
  _Float16* xh = (_Float16*)ws;  ws += (size_t)BB * TT * CC * 2;   // 12.6 MB
  _Float16* qh = (_Float16*)ws;  ws += (size_t)BB * TT * HH * 2;   // 1 MB
  _Float16* kh = (_Float16*)ws;  ws += (size_t)BB * TT * HH * 2;   // 1 MB
  _Float16* vT = (_Float16*)ws;  ws += (size_t)BB * HH * TT * 2;   // 1 MB
  _Float16* WT = (_Float16*)ws;  ws += 3L * HH * CC * 2;           // 288 KB

  const long nconv = (long)BB * TT * CC + 3L * HH * CC;
  int cblocks = (int)((nconv + 255) / 256);
  if (cblocks > 4096) cblocks = 4096;
  convert_kernel<<<cblocks, 256, 0, stream>>>(x, Wq, Wk, Wv, xh, WT);

  // 6144 output tiles (B * T/16 * 3 proj * 4 n-tiles), 8 waves per block
  proj_kernel<<<768, 256, 0, stream>>>(xh, WT, qh, kh, vT);

  // 512 query tiles (B * T/16), 4 waves per block
  attn_kernel<<<128, 128, 0, stream>>>(qh, kh, vT, out);
}
